// MultiHeadLinearAttention_20890720928061
// MI455X (gfx1250) — matrix operand order repaired and checked by reading, not compiled, not run on hardware
//
#include <hip/hip_runtime.h>
#include <hip/hip_bf16.h>

// ---------------------------------------------------------------------------
// MultiHeadLinearAttention (Linformer head-dim projection) for MI455X/gfx1250
// Algebraic folding:  kp = x @ (Wk^T E),  vp = x @ (Wv^T F)  =>
//   one fused QKV gemm, causal flash attention (d=64), output projection.
// All heavy contractions in f16 WMMA (v_wmma_f32_16x16x32_f16), fp32 accum.
// K tiles staged LDS via gfx1250 async load-to-LDS when available (ASYNCcnt).
// ---------------------------------------------------------------------------

typedef __attribute__((ext_vector_type(16))) _Float16 v16h;
typedef __attribute__((ext_vector_type(8)))  float    v8f;

#define B_DIM 2
#define T_DIM 2048
#define C_DIM 1024
#define H_DIM 16
#define HS    64
#define KD    64
#define M_ALL (B_DIM * T_DIM)      // 4096 rows (b,t)
#define N_ALL (H_DIM * 192)        // 3072 fused qkv cols

#if defined(__has_builtin)
#if __has_builtin(__builtin_amdgcn_global_load_async_to_lds_b128) && \
    __has_builtin(__builtin_amdgcn_s_wait_asynccnt)
#define HAVE_ASYNC_LDS 1
#endif
#endif

#ifdef HAVE_ASYNC_LDS
// ROCm clang-22 signature (probe: round-2 diagnostic): generic pointers to
// "int __vector(4)" for both the global source and the LDS destination.
typedef int v4i_vs __attribute__((vector_size(16)));
__device__ inline void async_copy_b128(const void* g, void* l) {
    __builtin_amdgcn_global_load_async_to_lds_b128(
        (v4i_vs*)(void*)g, (v4i_vs*)l, 0, 0);
}
#endif

union Frag16 {
    v16h v;
    uint4 q[2];
    _Float16 h[16];
};

__device__ inline v8f wmma_f16(v16h a, v16h b, v8f c) {
    return __builtin_amdgcn_wmma_f32_16x16x32_f16(
        /*neg_a=*/false, a, /*neg_b=*/false, b,
        /*c_mod=*/(short)0, c, /*reuse_a=*/false, /*reuse_b=*/false);
}

// A fragment (16x32 f16, MxK), rows row0..row0+15 from row-major [*, ld].
// lanes 0-15 row=l, K0..15; lanes 16-31 row=l-16, K16..31 (matches load_frag_b).
__device__ inline v16h load_frag_a(const _Float16* base, int row0, int ld, int k0) {
    int l  = threadIdx.x & 31;
    int hi = l >> 4;
    int r  = l & 15;
    const _Float16* p = base + (size_t)(row0 + r) * ld + k0 + hi * 16;
    Frag16 f;
    f.q[0] = *(const uint4*)(p);
    f.q[1] = *(const uint4*)(p + 8);
    return f.v;
}

// B fragment (32x16 f16, KxN) loaded from a TRANSPOSED store Bt[n][k] (row-major,
// leading dim ld) so each lane's 16 K-values are contiguous.
// lanes 0-15: col=l, K k0..k0+15; lanes 16-31: col=l-16, K k0+16..k0+31.
__device__ inline v16h load_frag_b(const _Float16* baseT, int col0, int ld, int k0) {
    int l  = threadIdx.x & 31;
    int hi = l >> 4;
    int c  = l & 15;
    const _Float16* p = baseT + (size_t)(col0 + c) * ld + k0 + hi * 16;
    Frag16 f;
    f.q[0] = *(const uint4*)(p);
    f.q[1] = *(const uint4*)(p + 8);
    return f.v;
}

// ---------------------------------------------------------------------------
// Kernel 0: fp32 -> f16 conversion (vectorized x4)
// ---------------------------------------------------------------------------
__global__ void cvt_f16_kernel(const float* __restrict__ src,
                               _Float16* __restrict__ dst, int n) {
    int i = (blockIdx.x * blockDim.x + threadIdx.x) * 4;
    if (i < n) {
        float4 v = *(const float4*)(src + i);
        dst[i + 0] = (_Float16)v.x;
        dst[i + 1] = (_Float16)v.y;
        dst[i + 2] = (_Float16)v.z;
        dst[i + 3] = (_Float16)v.w;
    }
}

// ---------------------------------------------------------------------------
// Kernel 1: build fused transposed weights Wall_t[h][j][c], j in [0,192):
//   j<64   : Wq[h][j][c] * 0.125          (softmax scale folded in)
//   64..127: sum_d Wk[h][d][c] * E[h][d][j-64]
//   128..  : sum_d Wv[h][d][c] * F[h][d][j-128]
// ---------------------------------------------------------------------------
__global__ void make_wall_kernel(const float* __restrict__ Wq,
                                 const float* __restrict__ Wk,
                                 const float* __restrict__ Wv,
                                 const float* __restrict__ E,
                                 const float* __restrict__ F,
                                 _Float16* __restrict__ Wall) {
    int idx = blockIdx.x * blockDim.x + threadIdx.x;   // h*192*1024 + j*1024 + c
    int c = idx & (C_DIM - 1);
    int hj = idx >> 10;
    int j = hj % 192;
    int h = hj / 192;
    float val;
    if (j < 64) {
        val = Wq[((size_t)h * HS + j) * C_DIM + c] * 0.125f;  // 1/sqrt(64)
    } else if (j < 128) {
        int k = j - 64;
        float s = 0.f;
#pragma unroll 8
        for (int d = 0; d < HS; ++d)
            s += Wk[((size_t)h * HS + d) * C_DIM + c] * E[((size_t)h * HS + d) * KD + k];
        val = s;
    } else {
        int k = j - 128;
        float s = 0.f;
#pragma unroll 8
        for (int d = 0; d < HS; ++d)
            s += Wv[((size_t)h * HS + d) * C_DIM + c] * F[((size_t)h * HS + d) * KD + k];
        val = s;
    }
    Wall[(size_t)idx] = (_Float16)val;
}

// ---------------------------------------------------------------------------
// Kernel 2: fused QKV gemm.  [4096 x 1024] x per-head [1024 x 192] -> Q/Kp/Vp
// One 16x64 tile per wave (A fragment reused across 4 WMMAs per k-step).
// n-group is always 64-aligned so a tile maps to exactly one of Q/Kp/Vp.
// ---------------------------------------------------------------------------
__global__ void __launch_bounds__(256) qkv_gemm_kernel(const _Float16* __restrict__ xh,
                                                       const _Float16* __restrict__ Wall,
                                                       _Float16* __restrict__ Q,
                                                       _Float16* __restrict__ Kp,
                                                       _Float16* __restrict__ Vp) {
    int wave  = threadIdx.x >> 5;
    int tile  = blockIdx.x * 8 + wave;            // 12288 tiles total
    int ntile = tile % (N_ALL / 64);              // 48
    int mtile = tile / (N_ALL / 64);              // 256
    int n0 = ntile * 64, m0 = mtile * 16;
    int h  = n0 / 192;
    int jg = n0 % 192;                            // 0, 64 or 128
    const _Float16* Bt = Wall + (size_t)h * 192 * C_DIM;

    v8f a0 = {}, a1 = {}, a2 = {}, a3 = {};
#pragma unroll 2
    for (int k0 = 0; k0 < C_DIM; k0 += 32) {
        v16h a = load_frag_a(xh, m0, C_DIM, k0);
        a0 = wmma_f16(a, load_frag_b(Bt, jg + 0,  C_DIM, k0), a0);
        a1 = wmma_f16(a, load_frag_b(Bt, jg + 16, C_DIM, k0), a1);
        a2 = wmma_f16(a, load_frag_b(Bt, jg + 32, C_DIM, k0), a2);
        a3 = wmma_f16(a, load_frag_b(Bt, jg + 48, C_DIM, k0), a3);
    }

    _Float16* dst = (jg == 0) ? Q : (jg == 64) ? Kp : Vp;

    int l  = threadIdx.x & 31;
    int cc = l & 15;
    int rb = (l >> 4) * 8;
#pragma unroll
    for (int r = 0; r < 8; ++r) {
        int m = m0 + rb + r;
        int b = m >> 11;           // /T_DIM
        int t = m & (T_DIM - 1);
        size_t base = ((((size_t)b * H_DIM) + h) * T_DIM + t) * 64;
        dst[base + 0  + cc] = (_Float16)a0[r];
        dst[base + 16 + cc] = (_Float16)a1[r];
        dst[base + 32 + cc] = (_Float16)a2[r];
        dst[base + 48 + cc] = (_Float16)a3[r];
    }
}

// ---------------------------------------------------------------------------
// Kernel 3: causal flash attention per (b,h).  Block = 4 waves = 64 q rows;
// s-steps of 32.  K staged to LDS via async load-to-LDS (ASYNCcnt path),
// V staged TRANSPOSED so P@V B-fragments are contiguous ds_load_b128.
// Writes Oa in (B, T, H*64) layout for the final projection.
// ---------------------------------------------------------------------------
__global__ void __launch_bounds__(128) flash_attn_kernel(const _Float16* __restrict__ Q,
                                                         const _Float16* __restrict__ Kp,
                                                         const _Float16* __restrict__ Vp,
                                                         _Float16* __restrict__ Oa) {
    __shared__ _Float16 Kt[32 * 64];      // (s, d) row-major
    __shared__ _Float16 VtT[64 * 32];     // (d, s) transposed
    __shared__ _Float16 Pl[4][16 * 32];

    int bh = blockIdx.y;                 // b*16 + h
    int i0 = blockIdx.x * 64;
    int w  = threadIdx.x >> 5;
    int l  = threadIdx.x & 31;
    int cc = l & 15;
    int rb = (l >> 4) * 8;
    int tid = threadIdx.x;

    const _Float16* Qh = Q  + (size_t)bh * T_DIM * 64;
    const _Float16* Kh = Kp + (size_t)bh * T_DIM * 64;
    const _Float16* Vh = Vp + (size_t)bh * T_DIM * 64;

    v16h qa0 = load_frag_a(Qh, i0 + w * 16, 64, 0);
    v16h qa1 = load_frag_a(Qh, i0 + w * 16, 64, 32);

    v8f o0 = {}, o1 = {}, o2 = {}, o3 = {};
    float mrun[8], lrun[8];
#pragma unroll
    for (int r = 0; r < 8; ++r) { mrun[r] = -1e30f; lrun[r] = 0.f; }

    int nsteps = i0 / 32 + 2;
    for (int st = 0; st < nsteps; ++st) {
        int s0 = st * 32;
        // ---- K tile: verbatim 4KB copy, async direct-to-LDS when available
        {
            const _Float16* gk = Kh + (size_t)s0 * 64;
#ifdef HAVE_ASYNC_LDS
            async_copy_b128(gk + (size_t)tid * 8,         Kt + (size_t)tid * 8);
            async_copy_b128(gk + (size_t)(tid + 128) * 8, Kt + (size_t)(tid + 128) * 8);
#else
            uint4* sk = (uint4*)Kt;
            const uint4* g4 = (const uint4*)gk;
            sk[tid]       = g4[tid];
            sk[tid + 128] = g4[tid + 128];
#endif
        }
        // ---- V tile: transpose while staging (8 halves per uint4 scatter)
        {
            const _Float16* gv = Vh + (size_t)s0 * 64;
#pragma unroll
            for (int it = 0; it < 2; ++it) {
                int idx = tid + it * 128;
                int dg = idx >> 5;         // 0..7 (group of 8 d values)
                int ss = idx & 31;         // s within tile
                union { uint4 u; _Float16 h[8]; } uu;
                uu.u = *(const uint4*)(gv + (size_t)ss * 64 + dg * 8);
#pragma unroll
                for (int j = 0; j < 8; ++j)
                    VtT[(dg * 8 + j) * 32 + ss] = uu.h[j];
            }
        }
#ifdef HAVE_ASYNC_LDS
        __builtin_amdgcn_s_wait_asynccnt(0);
#endif
        __syncthreads();
        if (st + 1 < nsteps) {   // gfx1250 global_prefetch_b8 of next K/V tiles
            __builtin_prefetch(Kh + (size_t)(s0 + 32) * 64, 0, 1);
            __builtin_prefetch(Vh + (size_t)(s0 + 32) * 64, 0, 1);
        }

        // ---- scores S = Q (16x64) . Kp^T : two 16x16 tiles over s ----
        v8f sv0 = {}, sv1 = {};
        sv0 = wmma_f16(qa0, load_frag_b(Kt, 0,  64, 0),  sv0);
        sv0 = wmma_f16(qa1, load_frag_b(Kt, 0,  64, 32), sv0);
        sv1 = wmma_f16(qa0, load_frag_b(Kt, 16, 64, 0),  sv1);
        sv1 = wmma_f16(qa1, load_frag_b(Kt, 16, 64, 32), sv1);

        // ---- causal mask + online softmax (rows live in 16-lane groups) ----
#pragma unroll
        for (int r = 0; r < 8; ++r) {
            int row = i0 + w * 16 + rb + r;
            float a = sv0[r];
            float b = sv1[r];
            if (s0 + cc      > row) a = -1e30f;
            if (s0 + 16 + cc > row) b = -1e30f;
            float mx = fmaxf(a, b);
            mx = fmaxf(mx, __shfl_xor(mx, 8, 32));
            mx = fmaxf(mx, __shfl_xor(mx, 4, 32));
            mx = fmaxf(mx, __shfl_xor(mx, 2, 32));
            mx = fmaxf(mx, __shfl_xor(mx, 1, 32));
            float mnew  = fmaxf(mrun[r], mx);
            float alpha = __expf(mrun[r] - mnew);
            float pa = __expf(a - mnew);
            float pb = __expf(b - mnew);
            float rs = pa + pb;
            rs += __shfl_xor(rs, 8, 32);
            rs += __shfl_xor(rs, 4, 32);
            rs += __shfl_xor(rs, 2, 32);
            rs += __shfl_xor(rs, 1, 32);
            lrun[r] = lrun[r] * alpha + rs;
            mrun[r] = mnew;
            o0[r] *= alpha; o1[r] *= alpha; o2[r] *= alpha; o3[r] *= alpha;
            // stage P (f16) into per-wave LDS for re-layout as A fragment
            Pl[w][(rb + r) * 32 + cc]      = (_Float16)pa;
            Pl[w][(rb + r) * 32 + 16 + cc] = (_Float16)pb;
        }

        // ---- O += P (16x32) @ V (32x64), V fragments contiguous from VtT ----
        v16h pf = load_frag_a(&Pl[w][0], 0, 32, 0);
        o0 = wmma_f16(pf, load_frag_b(VtT, 0,  32, 0), o0);
        o1 = wmma_f16(pf, load_frag_b(VtT, 16, 32, 0), o1);
        o2 = wmma_f16(pf, load_frag_b(VtT, 32, 32, 0), o2);
        o3 = wmma_f16(pf, load_frag_b(VtT, 48, 32, 0), o3);
        __syncthreads();
    }

    // epilogue: normalize and write Oa in (B, T, H*64) layout
    int b = bh >> 4;
    int h = bh & 15;
#pragma unroll
    for (int r = 0; r < 8; ++r) {
        float inv = 1.0f / lrun[r];
        size_t orow = ((size_t)b * T_DIM + (i0 + w * 16 + rb + r)) * C_DIM + h * 64;
        Oa[orow + cc]      = (_Float16)(o0[r] * inv);
        Oa[orow + 16 + cc] = (_Float16)(o1[r] * inv);
        Oa[orow + 32 + cc] = (_Float16)(o2[r] * inv);
        Oa[orow + 48 + cc] = (_Float16)(o3[r] * inv);
    }
}

// ---------------------------------------------------------------------------
// Kernel 4: output projection  out = Oa @ Wp^T + bp   (4096 x 1024 x 1024)
// Wp is (n, c) row-major so its rows ARE the transposed-B contiguous K runs.
// 16x64 tile per wave, A fragment reused across 4 WMMAs.
// ---------------------------------------------------------------------------
__global__ void __launch_bounds__(256) out_proj_kernel(const _Float16* __restrict__ Oa,
                                                       const _Float16* __restrict__ Wph,
                                                       const float* __restrict__ bp,
                                                       float* __restrict__ out) {
    int wave  = threadIdx.x >> 5;
    int tile  = blockIdx.x * 8 + wave;            // 4096 tiles
    int ntile = tile % (C_DIM / 64);              // 16
    int mtile = tile / (C_DIM / 64);              // 256
    int n0 = ntile * 64, m0 = mtile * 16;

    v8f a0 = {}, a1 = {}, a2 = {}, a3 = {};
#pragma unroll 2
    for (int k0 = 0; k0 < C_DIM; k0 += 32) {
        v16h a = load_frag_a(Oa, m0, C_DIM, k0);
        a0 = wmma_f16(a, load_frag_b(Wph, n0 + 0,  C_DIM, k0), a0);
        a1 = wmma_f16(a, load_frag_b(Wph, n0 + 16, C_DIM, k0), a1);
        a2 = wmma_f16(a, load_frag_b(Wph, n0 + 32, C_DIM, k0), a2);
        a3 = wmma_f16(a, load_frag_b(Wph, n0 + 48, C_DIM, k0), a3);
    }

    int l  = threadIdx.x & 31;
    int cc = l & 15;
    int rb = (l >> 4) * 8;
    float b0 = bp[n0 + cc];
    float b1 = bp[n0 + 16 + cc];
    float b2 = bp[n0 + 32 + cc];
    float b3 = bp[n0 + 48 + cc];
#pragma unroll
    for (int r = 0; r < 8; ++r) {
        int m = m0 + rb + r;
        float* po = out + (size_t)m * C_DIM + n0;
        po[cc]      = a0[r] + b0;
        po[16 + cc] = a1[r] + b1;
        po[32 + cc] = a2[r] + b2;
        po[48 + cc] = a3[r] + b3;
    }
}

// ---------------------------------------------------------------------------
extern "C" void kernel_launch(void* const* d_in, const int* in_sizes, int n_in,
                              void* d_out, int out_size, void* d_ws, size_t ws_size,
                              hipStream_t stream) {
    const float* x  = (const float*)d_in[0];
    const float* Wq = (const float*)d_in[1];
    const float* Wk = (const float*)d_in[2];
    const float* Wv = (const float*)d_in[3];
    const float* E  = (const float*)d_in[4];
    const float* F  = (const float*)d_in[5];
    const float* Wp = (const float*)d_in[6];
    const float* bp = (const float*)d_in[7];
    float* out = (float*)d_out;

    char* w = (char*)d_ws;
    size_t o = 0;
    _Float16* xh   = (_Float16*)(w + o); o += (size_t)M_ALL * C_DIM * 2;          // 8 MB
    _Float16* Wall = (_Float16*)(w + o); o += (size_t)H_DIM * 192 * C_DIM * 2;    // 6 MB
    _Float16* Wph  = (_Float16*)(w + o); o += (size_t)C_DIM * C_DIM * 2;          // 2 MB
    _Float16* Qb   = (_Float16*)(w + o); o += (size_t)B_DIM * H_DIM * T_DIM * 64 * 2; // 8 MB
    _Float16* Kpb  = (_Float16*)(w + o); o += (size_t)B_DIM * H_DIM * T_DIM * 64 * 2; // 8 MB
    _Float16* Vpb  = (_Float16*)(w + o); o += (size_t)B_DIM * H_DIM * T_DIM * 64 * 2; // 8 MB
    _Float16* Oa   = (_Float16*)(w + o); o += (size_t)M_ALL * C_DIM * 2;          // 8 MB

    // 0) fp32 -> f16 conversions
    cvt_f16_kernel<<<(M_ALL * C_DIM / 4 + 255) / 256, 256, 0, stream>>>(x, xh, M_ALL * C_DIM);
    cvt_f16_kernel<<<(C_DIM * C_DIM / 4 + 255) / 256, 256, 0, stream>>>(Wp, Wph, C_DIM * C_DIM);

    // 1) fold E/F into Wk/Wv, transpose weights, fold softmax scale into Wq
    make_wall_kernel<<<(H_DIM * 192 * C_DIM) / 256, 256, 0, stream>>>(Wq, Wk, Wv, E, F, Wall);

    // 2) fused QKV gemm (WMMA, 16x64 wave tiles)
    qkv_gemm_kernel<<<(M_ALL / 16) * (N_ALL / 64) / 8, 256, 0, stream>>>(xh, Wall, Qb, Kpb, Vpb);

    // 3) causal flash attention (WMMA + online softmax + async LDS staging)
    dim3 ag(T_DIM / 64, B_DIM * H_DIM);
    flash_attn_kernel<<<ag, 128, 0, stream>>>(Qb, Kpb, Vpb, Oa);

    // 4) output projection + bias (WMMA, 16x64 wave tiles)
    out_proj_kernel<<<(M_ALL / 16) * (C_DIM / 64) / 8, 256, 0, stream>>>(Oa, Wph, bp, out);
}
